// SpeculativeCrossLayerMoEModel_64141041598881
// MI455X (gfx1250) — matrix operand order repaired and checked by reading, not compiled, not run on hardware
//
#include <hip/hip_runtime.h>
#include <hip/hip_bf16.h>
#include <cstdint>
#include <cstddef>

typedef __attribute__((ext_vector_type(16))) _Float16 v16h;
typedef __attribute__((ext_vector_type(8)))  _Float16 v8h;
typedef __attribute__((ext_vector_type(8)))  float    v8f;

namespace {

constexpr int Bc = 128, Tc = 32, Cc = 128, Hc = 4, HSc = 32, Ec = 4, Lc = 4;
constexpr int Vc = 50257, VPAD = 50432;          // VPAD multiple of 256 (64*NB)
constexpr int NT = Bc * Tc;                      // 4096 tokens
constexpr int C4 = 4 * Cc;                       // 512
constexpr float LN_EPS    = 1e-5f;
constexpr float ATT_SCALE = 0.08838834764831845f; // C^-0.5 (full C per reference)

// ---------------------------------------------------------------------------
// Embedding: x[b,t,c] = tok_emb[idx[b,t], c] + pos_emb[t, c]
// ---------------------------------------------------------------------------
__global__ void embed_kern(const int* __restrict__ idx,
                           const float* __restrict__ tok,
                           const float* __restrict__ pos,
                           float* __restrict__ x) {
  int i = blockIdx.x * blockDim.x + threadIdx.x;
  if (i >= NT * Cc) return;
  int t = i / Cc, c = i % Cc;
  int tt = t % Tc;
  x[i] = tok[(size_t)idx[t] * Cc + c] + pos[tt * Cc + c];
}

// ---------------------------------------------------------------------------
// Weight packing: W[K,N] f32 row-major -> Wt[Npad,K] f16 (transposed, padded)
// ---------------------------------------------------------------------------
__global__ void pack2d_kern(const float* __restrict__ W, _Float16* __restrict__ Wt,
                            int K, int N, long total) {
  long i = (long)blockIdx.x * blockDim.x + threadIdx.x;
  if (i >= total) return;
  int n = (int)(i / K), k = (int)(i % K);
  Wt[i] = (n < N) ? (_Float16)W[(size_t)k * N + n] : (_Float16)0.f;
}

// Wq/Wk/Wv [H,C,HS] -> Wt[(h*HS+d), c] f16, so Q = h @ Wt^T gives [.., (h,d)]
__global__ void pack_qkv_kern(const float* __restrict__ W, _Float16* __restrict__ Wt) {
  int i = blockIdx.x * blockDim.x + threadIdx.x;
  if (i >= Cc * Cc) return;
  int n = i >> 7, k = i & 127;          // n = h*HS+d (col), k = c (row)
  int h = n >> 5, d = n & 31;
  Wt[i] = (_Float16)W[((size_t)h * Cc + k) * HSc + d];
}

// ---------------------------------------------------------------------------
// f32 -> f16 conversion
// ---------------------------------------------------------------------------
__global__ void cvt_f16_kern(const float* __restrict__ in, _Float16* __restrict__ out, int n) {
  int i = blockIdx.x * blockDim.x + threadIdx.x;
  if (i < n) out[i] = (_Float16)in[i];
}

// ---------------------------------------------------------------------------
// LayerNorm: one wave per token (wave32), lane handles 4 channels
// ---------------------------------------------------------------------------
__global__ void ln_kern(const float* __restrict__ x,
                        const float* __restrict__ g, const float* __restrict__ b,
                        float* __restrict__ y32, _Float16* __restrict__ y16) {
  int tok  = blockIdx.x * (blockDim.x >> 5) + (threadIdx.x >> 5);
  int lane = threadIdx.x & 31;
  if (tok >= NT) return;
  const float* xr = x + (size_t)tok * Cc;
  float v[4], s = 0.f, s2 = 0.f;
#pragma unroll
  for (int i = 0; i < 4; ++i) {
    v[i] = xr[i * 32 + lane];
    s += v[i]; s2 += v[i] * v[i];
  }
#pragma unroll
  for (int o = 16; o > 0; o >>= 1) {
    s  += __shfl_xor(s,  o, 32);
    s2 += __shfl_xor(s2, o, 32);
  }
  float mean = s * (1.f / Cc);
  float var  = s2 * (1.f / Cc) - mean * mean;
  float inv  = rsqrtf(var + LN_EPS);
#pragma unroll
  for (int i = 0; i < 4; ++i) {
    int c = i * 32 + lane;
    float y = (v[i] - mean) * inv * g[c] + b[c];
    if (y32) y32[(size_t)tok * Cc + c] = y;
    if (y16) y16[(size_t)tok * Cc + c] = (_Float16)y;
  }
}

// ---------------------------------------------------------------------------
// Register-blocked wave-tile WMMA GEMM:
//   D = A[M,K](f16) x Bt[Npad,K](f16)^T (+bias)
// Each wave computes one 16-row M tile x NB consecutive 16-col N tiles,
// reusing the A fragment across NB wmmas (raises L2 arithmetic intensity).
//   grid = (Npad/(64*NB), M/16), block = 128 (4 waves)
//   MODE 0: C[m*ldc+n] = acc+bias                (store, guarded n<N)
//   MODE 1: Ch[m*N+n]  = f16(relu(acc+bias))     (expert GEMM1)
//   MODE 2: C[m*ldc+n] += acc+bias               (residual accumulate)
//   MODE 3: C[m*ldc+n] += rowScale[m*rsStride]*(acc+bias)  (gated MoE acc)
//   MODE 4: like 0 but nontemporal store         (huge streaming LM logits)
// A-frag lane: row=lane&15, halves [k0+ks*8,+8) and [k0+ks*8+16,+8), ks=lane>>4
// B-frag lane: col=lane&15, halves [k0+ks*8,+8) and [k0+ks*8+16,+8)
// C/D   lane: col=lane&15, VGPR r -> row (ks*8 + r)
// ---------------------------------------------------------------------------
template<int MODE, int NB>
__global__ void gemm_wmma_kern(const _Float16* __restrict__ A,
                               const _Float16* __restrict__ Bt,
                               const float* __restrict__ bias,
                               float* __restrict__ C,
                               _Float16* __restrict__ Ch,
                               const float* __restrict__ rowScale, int rsStride,
                               int M, int N, int K, int ldc) {
  int lane   = threadIdx.x & 31;
  int wave   = threadIdx.x >> 5;
  int tileN0 = (blockIdx.x * 4 + wave) * NB;
  int tileM  = blockIdx.y;
  int ncol   = lane & 15;
  int ksel   = lane >> 4;
  int mrow   = tileM * 16 + (lane & 15);

  v8f acc[NB];
  float bval[NB];
#pragma unroll
  for (int j = 0; j < NB; ++j) {
    acc[j] = (v8f){0.f, 0.f, 0.f, 0.f, 0.f, 0.f, 0.f, 0.f};
    int nj = (tileN0 + j) * 16 + ncol;
    bval[j] = (bias && nj < N) ? bias[nj] : 0.f;
  }

  const _Float16* aptr = A  + (size_t)mrow * K + ksel * 8;
  const _Float16* bptr = Bt + ((size_t)tileN0 * 16 + ncol) * K + ksel * 8;

  for (int k0 = 0; k0 < K; k0 += 32) {
    v8h alo = *(const v8h*)(aptr + k0);
    v8h ahi = *(const v8h*)(aptr + k0 + 16);
    v16h a;
#pragma unroll
    for (int i = 0; i < 8; ++i) { a[i] = alo[i]; a[i + 8] = ahi[i]; }
#pragma unroll
    for (int j = 0; j < NB; ++j) {
      const _Float16* bj = bptr + (size_t)j * 16 * K + k0;
      v8h blo = *(const v8h*)(bj);
      v8h bhi = *(const v8h*)(bj + 16);
      v16h bb;
#pragma unroll
      for (int i = 0; i < 8; ++i) { bb[i] = blo[i]; bb[i + 8] = bhi[i]; }
      acc[j] = __builtin_amdgcn_wmma_f32_16x16x32_f16(
          false, a, false, bb, (short)0, acc[j], false, false);
    }
  }

  int moff = ksel * 8;
#pragma unroll
  for (int j = 0; j < NB; ++j) {
    int n = (tileN0 + j) * 16 + ncol;
#pragma unroll
    for (int r = 0; r < 8; ++r) {
      int m = tileM * 16 + moff + r;
      float val = acc[j][r] + bval[j];
      if (MODE == 0) {
        if (n < N) C[(size_t)m * ldc + n] = val;
      } else if (MODE == 1) {
        Ch[(size_t)m * N + n] = (_Float16)fmaxf(val, 0.f);
      } else if (MODE == 2) {
        C[(size_t)m * ldc + n] += val;
      } else if (MODE == 3) {
        C[(size_t)m * ldc + n] += rowScale[(size_t)m * rsStride] * val;
      } else { // MODE 4: streaming nontemporal store (don't pollute L2)
        if (n < N) __builtin_nontemporal_store(val, &C[(size_t)m * ldc + n]);
      }
    }
  }
}

// ---------------------------------------------------------------------------
// Causal attention, one block (32 threads) per (b, head). T=HS=32.
// q/k/v/o layout: [NT, C] with head h occupying columns [h*HS, (h+1)*HS)
// ---------------------------------------------------------------------------
__global__ void attention_kern(const float* __restrict__ q,
                               const float* __restrict__ k,
                               const float* __restrict__ v,
                               float* __restrict__ o) {
  __shared__ float ks[32][33];
  __shared__ float vs[32][33];
  __shared__ float ss[32][33];
  int bh = blockIdx.x;
  int b  = bh / Hc, hh = bh % Hc;
  int t  = threadIdx.x;
  int col0 = hh * HSc;

  for (int r = 0; r < 32; ++r) {
    ks[r][t] = k[((size_t)(b * Tc + r)) * Cc + col0 + t];
    vs[r][t] = v[((size_t)(b * Tc + r)) * Cc + col0 + t];
  }
  __syncthreads();

  float qv[32];
#pragma unroll
  for (int d = 0; d < 32; ++d)
    qv[d] = q[((size_t)(b * Tc + t)) * Cc + col0 + d];

  float mx = -1e30f;
  for (int s = 0; s < 32; ++s) {
    float d = 0.f;
#pragma unroll
    for (int dd = 0; dd < 32; ++dd) d += qv[dd] * ks[s][dd];
    d = (s <= t) ? d * ATT_SCALE : -1e30f;
    ss[t][s] = d;
    mx = fmaxf(mx, d);
  }
  float den = 0.f;
  for (int s = 0; s < 32; ++s) {
    float w = __expf(ss[t][s] - mx);
    ss[t][s] = w;
    den += w;
  }
  float inv = 1.f / den;
  float out[32];
#pragma unroll
  for (int dd = 0; dd < 32; ++dd) out[dd] = 0.f;
  for (int s = 0; s <= t; ++s) {
    float w = ss[t][s] * inv;
#pragma unroll
    for (int dd = 0; dd < 32; ++dd) out[dd] += w * vs[s][dd];
  }
#pragma unroll
  for (int dd = 0; dd < 32; ++dd)
    o[((size_t)(b * Tc + t)) * Cc + col0 + dd] = out[dd];
}

// ---------------------------------------------------------------------------
// Router: per-token noisy top-2 gating + skip gate; outputs effective gates
// (skip folded in: skip -> all-zero gates -> x unchanged).
// eps noise: deterministic hash-based gaussian (JAX PRNG not reproducible).
// ---------------------------------------------------------------------------
__device__ __forceinline__ uint32_t hashu(uint32_t x) {
  x ^= x >> 16; x *= 0x7feb352du;
  x ^= x >> 15; x *= 0x846ca68bu;
  x ^= x >> 16; return x;
}
__device__ __forceinline__ float hash_normal(uint32_t seed) {
  uint32_t a = hashu(seed), b = hashu(seed ^ 0x9e3779b9u);
  float u1 = ((float)a + 1.0f) * 2.3283064e-10f;   // (0,1]
  float u2 = (float)b * 2.3283064e-10f;
  return sqrtf(-2.f * logf(u1)) * cosf(6.2831853071795864f * u2);
}

__global__ void router_kern(const float* __restrict__ nx,
                            const float* __restrict__ Wr, const float* __restrict__ br,
                            const float* __restrict__ Wn, const float* __restrict__ bn,
                            const float* __restrict__ Ws, const float* __restrict__ bs,
                            float* __restrict__ gates, int layer) {
  int tok = blockIdx.x * blockDim.x + threadIdx.x;
  if (tok >= NT) return;
  const float* xr = nx + (size_t)tok * Cc;

  float logit[Ec], nz[Ec];
#pragma unroll
  for (int e = 0; e < Ec; ++e) { logit[e] = br[e]; nz[e] = bn[e]; }
  float sd = bs[0];
  for (int c = 0; c < Cc; ++c) {
    float xv = xr[c];
#pragma unroll
    for (int e = 0; e < Ec; ++e) {
      logit[e] += xv * Wr[c * Ec + e];
      nz[e]    += xv * Wn[c * Ec + e];
    }
    sd += xv * Ws[c];
  }
  float noisy[Ec];
#pragma unroll
  for (int e = 0; e < Ec; ++e) {
    float z  = nz[e];
    float sp = (z > 20.f) ? z : log1pf(expf(z));     // softplus
    float eps = hash_normal((uint32_t)(((layer * NT + tok) * Ec + e) * 2654435761u + 12345u));
    noisy[e] = logit[e] + eps * sp;
  }
  // top-2 of 4 (ties -> lowest index, matching lax.top_k)
  int i1 = 0;
#pragma unroll
  for (int e = 1; e < Ec; ++e) if (noisy[e] > noisy[i1]) i1 = e;
  int i2 = -1;
#pragma unroll
  for (int e = 0; e < Ec; ++e)
    if (e != i1 && (i2 < 0 || noisy[e] > noisy[i2])) i2 = e;

  bool skip = sd > 0.f;                               // sigmoid(sd) > 0.5
  float m  = noisy[i1];
  float e2 = __expf(noisy[i2] - m);
  float dn = 1.f + e2;
  float g1 = skip ? 0.f : 1.f / dn;
  float g2 = skip ? 0.f : e2 / dn;
#pragma unroll
  for (int e = 0; e < Ec; ++e) {
    float g = (e == i1) ? g1 : ((e == i2) ? g2 : 0.f);
    gates[(size_t)tok * Ec + e] = g;
  }
}

} // anonymous namespace

// ---------------------------------------------------------------------------
// Host orchestration
// ---------------------------------------------------------------------------
extern "C" void kernel_launch(void* const* d_in, const int* in_sizes, int n_in,
                              void* d_out, int out_size, void* d_ws, size_t ws_size,
                              hipStream_t stream) {
  const int*   idx     = (const int*)  d_in[0];
  const float* tok_emb = (const float*)d_in[1];
  const float* pos_emb = (const float*)d_in[2];
  const float* ln1_g   = (const float*)d_in[3];
  const float* ln1_b   = (const float*)d_in[4];
  const float* Wq      = (const float*)d_in[5];
  const float* Wk      = (const float*)d_in[6];
  const float* Wv      = (const float*)d_in[7];
  const float* Wo      = (const float*)d_in[8];
  const float* bo      = (const float*)d_in[9];
  const float* ln2_g   = (const float*)d_in[10];
  const float* ln2_b   = (const float*)d_in[11];
  const float* Wr      = (const float*)d_in[12];
  const float* br      = (const float*)d_in[13];
  const float* Wn      = (const float*)d_in[14];
  const float* bn      = (const float*)d_in[15];
  const float* Ws      = (const float*)d_in[16];
  const float* bs      = (const float*)d_in[17];
  const float* W1      = (const float*)d_in[18];
  const float* b1      = (const float*)d_in[19];
  const float* W2      = (const float*)d_in[20];
  const float* b2      = (const float*)d_in[21];
  const float* lnf_g   = (const float*)d_in[22];
  const float* lnf_b   = (const float*)d_in[23];
  const float* W_lm    = (const float*)d_in[24];
  const float* b_lm    = (const float*)d_in[25];
  float* out = (float*)d_out;

  // --- bump allocator over d_ws ---
  size_t off = 0;
  auto alloc = [&](size_t bytes) -> void* {
    off = (off + 255) & ~(size_t)255;
    void* p = (char*)d_ws + off;
    off += bytes;
    return p;
  };
  float*     x32   = (float*)    alloc((size_t)NT * Cc * 4);
  _Float16*  h16   = (_Float16*) alloc((size_t)NT * Cc * 2);
  float*     nx32  = (float*)    alloc((size_t)NT * Cc * 4);
  _Float16*  nx16  = (_Float16*) alloc((size_t)NT * Cc * 2);
  float*     q32   = (float*)    alloc((size_t)NT * Cc * 4);
  float*     k32   = (float*)    alloc((size_t)NT * Cc * 4);
  float*     v32   = (float*)    alloc((size_t)NT * Cc * 4);
  float*     o32   = (float*)    alloc((size_t)NT * Cc * 4);
  _Float16*  o16   = (_Float16*) alloc((size_t)NT * Cc * 2);
  _Float16*  h1f16 = (_Float16*) alloc((size_t)NT * C4 * 2);
  float*     gates = (float*)    alloc((size_t)NT * Ec * 4);
  _Float16*  wtq   = (_Float16*) alloc((size_t)Cc * Cc * 2);
  _Float16*  wtk   = (_Float16*) alloc((size_t)Cc * Cc * 2);
  _Float16*  wtv   = (_Float16*) alloc((size_t)Cc * Cc * 2);
  _Float16*  wto   = (_Float16*) alloc((size_t)Cc * Cc * 2);
  _Float16*  wt1   = (_Float16*) alloc((size_t)C4 * Cc * 2);
  _Float16*  wt2   = (_Float16*) alloc((size_t)Cc * C4 * 2);
  _Float16*  wtlm  = (_Float16*) alloc((size_t)VPAD * Cc * 2);
  (void)ws_size; (void)n_in; (void)in_sizes; (void)out_size;

  // templated GEMM launcher: grid covers Npad with 4 waves * NB tiles per block
  #define LAUNCH_GEMM(MODE, NB, A_, Bt_, bias_, C_, Ch_, rs_, rsStride_, M_, N_, Npad_, K_, ldc_) \
    hipLaunchKernelGGL((gemm_wmma_kern<MODE, NB>), dim3((Npad_) / (64 * (NB)), (M_) / 16), \
                       dim3(128), 0, stream, A_, Bt_, bias_, C_, Ch_, rs_, rsStride_, M_, N_, K_, ldc_)

  auto pack = [&](const float* W, _Float16* Wt, int K, int N, int Npad) {
    long total = (long)Npad * K;
    hipLaunchKernelGGL(pack2d_kern, dim3((total + 255) / 256), dim3(256), 0, stream,
                       W, Wt, K, N, total);
  };

  // 1) embedding
  hipLaunchKernelGGL(embed_kern, dim3((NT * Cc + 255) / 256), dim3(256), 0, stream,
                     idx, tok_emb, pos_emb, x32);

  // 2) pack LM head weights (once per call; deterministic)
  pack(W_lm, wtlm, Cc, Vc, VPAD);

  // 3) transformer blocks
  for (int l = 0; l < Lc; ++l) {
    // --- LN1 -> h16 ---
    hipLaunchKernelGGL(ln_kern, dim3(NT / 4), dim3(128), 0, stream,
                       x32, ln1_g + l * Cc, ln1_b + l * Cc, (float*)nullptr, h16);
    // --- QKV projections ---
    hipLaunchKernelGGL(pack_qkv_kern, dim3((Cc * Cc + 255) / 256), dim3(256), 0, stream,
                       Wq + (size_t)l * Hc * Cc * HSc, wtq);
    hipLaunchKernelGGL(pack_qkv_kern, dim3((Cc * Cc + 255) / 256), dim3(256), 0, stream,
                       Wk + (size_t)l * Hc * Cc * HSc, wtk);
    hipLaunchKernelGGL(pack_qkv_kern, dim3((Cc * Cc + 255) / 256), dim3(256), 0, stream,
                       Wv + (size_t)l * Hc * Cc * HSc, wtv);
    LAUNCH_GEMM(0, 2, h16, wtq, (const float*)nullptr, q32, (_Float16*)nullptr,
                (const float*)nullptr, 0, NT, Cc, Cc, Cc, Cc);
    LAUNCH_GEMM(0, 2, h16, wtk, (const float*)nullptr, k32, (_Float16*)nullptr,
                (const float*)nullptr, 0, NT, Cc, Cc, Cc, Cc);
    LAUNCH_GEMM(0, 2, h16, wtv, (const float*)nullptr, v32, (_Float16*)nullptr,
                (const float*)nullptr, 0, NT, Cc, Cc, Cc, Cc);
    // --- causal attention ---
    hipLaunchKernelGGL(attention_kern, dim3(Bc * Hc), dim3(32), 0, stream,
                       q32, k32, v32, o32);
    // --- output projection + residual: x += o @ Wo + bo ---
    hipLaunchKernelGGL(cvt_f16_kern, dim3((NT * Cc + 255) / 256), dim3(256), 0, stream,
                       o32, o16, NT * Cc);
    pack(Wo + (size_t)l * Cc * Cc, wto, Cc, Cc, Cc);
    LAUNCH_GEMM(2, 2, o16, wto, bo + l * Cc, x32, (_Float16*)nullptr,
                (const float*)nullptr, 0, NT, Cc, Cc, Cc, Cc);
    // --- LN2 -> nx (f32 for router, f16 for expert GEMMs) ---
    hipLaunchKernelGGL(ln_kern, dim3(NT / 4), dim3(128), 0, stream,
                       x32, ln2_g + l * Cc, ln2_b + l * Cc, nx32, nx16);
    // --- noisy top-2 router + skip gate -> effective gates ---
    hipLaunchKernelGGL(router_kern, dim3(NT / 128), dim3(128), 0, stream,
                       nx32, Wr + (size_t)l * Cc * Ec, br + (size_t)l * Ec,
                       Wn + (size_t)l * Cc * Ec, bn + (size_t)l * Ec,
                       Ws + (size_t)l * Cc, bs + l, gates, l);
    // --- experts: x += gate_e * (relu(nx@W1_e + b1_e) @ W2_e + b2_e) ---
    for (int e = 0; e < Ec; ++e) {
      size_t we = (size_t)(l * Ec + e);
      pack(W1 + we * Cc * C4, wt1, Cc, C4, C4);
      LAUNCH_GEMM(1, 4, nx16, wt1, b1 + we * C4, (float*)nullptr, h1f16,
                  (const float*)nullptr, 0, NT, C4, C4, Cc, C4);
      pack(W2 + we * C4 * Cc, wt2, C4, Cc, Cc);
      LAUNCH_GEMM(3, 2, h1f16, wt2, b2 + we * Cc, x32, (_Float16*)nullptr,
                  gates + e, Ec, NT, Cc, Cc, C4, Cc);
    }
  }

  // 4) final LN -> h16
  hipLaunchKernelGGL(ln_kern, dim3(NT / 4), dim3(128), 0, stream,
                     x32, lnf_g, lnf_b, (float*)nullptr, h16);

  // 5) LM head: out[NT, V] = h @ W_lm + b_lm  (nontemporal store, n<V guarded)
  LAUNCH_GEMM(4, 4, h16, wtlm, b_lm, out, (_Float16*)nullptr,
              (const float*)nullptr, 0, NT, Vc, VPAD, Cc, Vc);

  #undef LAUNCH_GEMM
}
